// NodeBlock_44865228374365
// MI455X (gfx1250) — compile-verified, matrix operand order checked by reading, not run on hardware
//
#include <hip/hip_runtime.h>

// ---------------------------------------------------------------------------
// Problem constants (from reference)
// ---------------------------------------------------------------------------
#define NNODES 100000
#define NEDGES 1600000
#define DE_ 48
#define DN_ 128
#define DG_ 64
#define KX 256   // DE+DN+DG = 240, padded to 256 for K%32==0
#define H_ 256
#define DO_ 128

#define M_TILES 2                       // 16-row M-tiles per wave (32 nodes)
#define WAVES_PER_BLOCK 2
#define TILES32 (NNODES / 32)           // 3125 exactly

// ---------------------------------------------------------------------------
// CDNA5 WMMA types
// ---------------------------------------------------------------------------
typedef __attribute__((ext_vector_type(16))) __bf16       v16bf;
typedef __attribute__((ext_vector_type(8)))  float        v8f;
typedef __attribute__((ext_vector_type(4)))  unsigned int u32x4;

union FragU {
    u32x4 q[2];
    v16bf v;
};

__device__ __forceinline__ unsigned short f2bf(float f) {
    // round-to-nearest-even fp32 -> bf16
    unsigned u = __float_as_uint(f);
    unsigned r = u + 0x7FFFu + ((u >> 16) & 1u);
    return (unsigned short)(r >> 16);
}

// ---------------------------------------------------------------------------
// CSR build, step 0: zero the per-node counters
// ---------------------------------------------------------------------------
__global__ void zero_u32_kernel(unsigned int* __restrict__ p, int n) {
    int i = blockIdx.x * blockDim.x + threadIdx.x;
    if (i < n) p[i] = 0u;
}

// ---------------------------------------------------------------------------
// CSR step 1: in-degree histogram (1.6M integer atomics)
// ---------------------------------------------------------------------------
__global__ void count_kernel(const int* __restrict__ receivers,
                             unsigned int* __restrict__ cnt) {
    int e = blockIdx.x * blockDim.x + threadIdx.x;
    if (e < NEDGES) atomicAdd(&cnt[receivers[e]], 1u);
}

// ---------------------------------------------------------------------------
// CSR step 2: exclusive prefix scan over 100K counts (single 1024-thread block)
// ---------------------------------------------------------------------------
__global__ void __launch_bounds__(1024)
scan_kernel(const unsigned int* __restrict__ cnt,
            unsigned int* __restrict__ offs,
            unsigned int* __restrict__ cursor) {
    __shared__ unsigned int partial[1024];
    const int T = 1024;
    const int t = threadIdx.x;
    const int chunk = (NNODES + T - 1) / T;      // 98
    const int lo = t * chunk;
    const int hi = (lo + chunk < NNODES) ? (lo + chunk) : NNODES;

    unsigned int s = 0;
    for (int i = lo; i < hi; ++i) s += cnt[i];
    partial[t] = s;
    __syncthreads();

    for (int d = 1; d < T; d <<= 1) {
        unsigned int v = (t >= d) ? partial[t - d] : 0u;
        __syncthreads();
        if (t >= d) partial[t] += v;
        __syncthreads();
    }

    unsigned int run = (t == 0) ? 0u : partial[t - 1];   // exclusive base
    for (int i = lo; i < hi; ++i) {
        unsigned int c = cnt[i];
        offs[i]   = run;
        cursor[i] = run;
        run += c;
    }
}

// ---------------------------------------------------------------------------
// CSR step 3: place edge indices into per-node buckets
// ---------------------------------------------------------------------------
__global__ void place_kernel(const int* __restrict__ receivers,
                             unsigned int* __restrict__ cursor,
                             unsigned int* __restrict__ eidx) {
    int e = blockIdx.x * blockDim.x + threadIdx.x;
    if (e < NEDGES) {
        int r = receivers[e];
        unsigned int p = atomicAdd(&cursor[r], 1u);
        eidx[p] = (unsigned int)e;
    }
}

// ---------------------------------------------------------------------------
// CSR step 4: gather-aggregate (NO atomics). One thread per (node, feature);
// 48 consecutive lanes stream each 192B edge row coalesced. Writes the MEAN
// directly into out[n][0..47] (later overwritten by the MLP).
// ---------------------------------------------------------------------------
__global__ void aggregate_mean_kernel(const float* __restrict__ edge_data,
                                      const unsigned int* __restrict__ offs,
                                      const unsigned int* __restrict__ cnt,
                                      const unsigned int* __restrict__ eidx,
                                      float* __restrict__ out) {
    int t = blockIdx.x * blockDim.x + threadIdx.x;   // NNODES*48
    if (t >= NNODES * DE_) return;
    int n = t / DE_;
    int f = t - n * DE_;
    unsigned int start = offs[n];
    unsigned int deg   = cnt[n];
    float s = 0.0f;
    for (unsigned int i = 0; i < deg; ++i) {
        int e = (int)eidx[start + i];
        s += edge_data[e * DE_ + f];
    }
    out[n * DO_ + f] = s / fmaxf((float)deg, 1.0f);
}

// ---------------------------------------------------------------------------
// Weight conversion fp32 -> bf16, transposed to [N][K] (K-contiguous) so each
// WMMA B-fragment is two contiguous 16B loads per lane.
// ---------------------------------------------------------------------------
__global__ void cvt_w1_kernel(const float* __restrict__ W1,
                              unsigned short* __restrict__ W1T) {
    int t = blockIdx.x * blockDim.x + threadIdx.x;   // 256*256
    int n = t >> 8;
    int k = t & 255;
    float v = (k < (DE_ + DN_ + DG_)) ? W1[k * H_ + n] : 0.0f;
    W1T[n * KX + k] = f2bf(v);
}

__global__ void cvt_w2_kernel(const float* __restrict__ W2,
                              unsigned short* __restrict__ W2T) {
    int t = blockIdx.x * blockDim.x + threadIdx.x;   // 128*256
    int n = t >> 8;
    int k = t & 255;
    W2T[n * H_ + k] = f2bf(W2[k * DO_ + n]);
}

// ---------------------------------------------------------------------------
// Fused MLP with WMMA. One wave per 32-node tile (2 M-sub-tiles), so each
// B-fragment load feeds two v_wmma_f32_16x16x32_bf16 ops.
// 16-bit A-fragment layout (ISA 7.12.2): lane = M%16, half = lane>>4;
//   VGPR0-3 hold K = koff+8*half .. +7   (16B contiguous),
//   VGPR4-7 hold K = koff+16+8*half .. +7 (16B contiguous).
// B mirrors A with lane = N (weights stored [N][K] row-major).
// ---------------------------------------------------------------------------
__global__ void __launch_bounds__(32 * WAVES_PER_BLOCK)
mlp_kernel(const float* __restrict__ node_data,
           const float* __restrict__ global_data,
           const float* __restrict__ b1,
           const float* __restrict__ b2,
           const unsigned short* __restrict__ W1T,
           const unsigned short* __restrict__ W2T,
           float* __restrict__ out) {
    __shared__ unsigned short xs[WAVES_PER_BLOCK][32 * KX];   // 16KB/wave
    __shared__ unsigned short hs[WAVES_PER_BLOCK][32 * H_];   // 16KB/wave

    const int tid  = threadIdx.x;
    const int wave = tid >> 5;
    const int lane = tid & 31;
    const int m16  = lane & 15;
    const int half = lane >> 4;

    const int tile = blockIdx.x * WAVES_PER_BLOCK + wave;
    if (tile >= TILES32) return;
    const int node0 = tile * 32;

    unsigned short* xw = xs[wave];
    unsigned short* hw = hs[wave];

    // ---- Build bf16 x-tile [32 x 256] in this wave's LDS slice ----
    for (int idx = lane; idx < 32 * KX; idx += 32) {
        const int m = idx >> 8;
        const int k = idx & (KX - 1);
        const int node = node0 + m;
        float v;
        if (k < DE_) {
            v = out[node * DO_ + k];                 // mean aggregate (pre-divided)
        } else if (k < DE_ + DN_) {
            v = node_data[node * DN_ + (k - DE_)];
        } else if (k < DE_ + DN_ + DG_) {
            v = global_data[k - (DE_ + DN_)];
        } else {
            v = 0.0f;                                // K padding
        }
        xw[idx] = f2bf(v);
    }

    // ---- Load A-fragments of x: 2 M-tiles x 8 K-chunks ----
    v16bf afrag[M_TILES][8];
#pragma unroll
    for (int t2 = 0; t2 < M_TILES; ++t2) {
#pragma unroll
        for (int kk = 0; kk < 8; ++kk) {
            const unsigned short* p =
                xw + (t2 * 16 + m16) * KX + kk * 32 + 8 * half;
            FragU f;
            f.q[0] = *(const u32x4*)(p);
            f.q[1] = *(const u32x4*)(p + 16);
            afrag[t2][kk] = f.v;
        }
    }

    // ---- GEMM1: 16 N-tiles x 8 K-steps x 2 M-tiles; bias+ReLU -> LDS bf16 ----
#pragma unroll 1
    for (int nt = 0; nt < 16; ++nt) {
        v8f acc0 = {};
        v8f acc1 = {};
        const unsigned short* wrow = W1T + (nt * 16 + m16) * KX;
#pragma unroll
        for (int kk = 0; kk < 8; ++kk) {
            const unsigned short* p = wrow + kk * 32 + 8 * half;
            FragU f;
            f.q[0] = *(const u32x4*)(p);
            f.q[1] = *(const u32x4*)(p + 16);
            acc0 = __builtin_amdgcn_wmma_f32_16x16x32_bf16(
                false, afrag[0][kk], false, f.v, (short)0, acc0, false, false);
            acc1 = __builtin_amdgcn_wmma_f32_16x16x32_bf16(
                false, afrag[1][kk], false, f.v, (short)0, acc1, false, false);
        }
        const float bias = b1[nt * 16 + m16];
#pragma unroll
        for (int r = 0; r < 8; ++r) {
            float h0 = fmaxf(acc0[r] + bias, 0.0f);
            float h1 = fmaxf(acc1[r] + bias, 0.0f);
            hw[(r + 8 * half) * H_ + nt * 16 + m16]      = f2bf(h0);
            hw[(16 + r + 8 * half) * H_ + nt * 16 + m16] = f2bf(h1);
        }
    }

    // ---- Reload A-fragments from h (layout transpose via LDS) ----
#pragma unroll
    for (int t2 = 0; t2 < M_TILES; ++t2) {
#pragma unroll
        for (int kk = 0; kk < 8; ++kk) {
            const unsigned short* p =
                hw + (t2 * 16 + m16) * H_ + kk * 32 + 8 * half;
            FragU f;
            f.q[0] = *(const u32x4*)(p);
            f.q[1] = *(const u32x4*)(p + 16);
            afrag[t2][kk] = f.v;
        }
    }

    // ---- GEMM2: 8 N-tiles x 8 K-steps x 2 M-tiles; bias; f32 stores ----
#pragma unroll 1
    for (int ot = 0; ot < 8; ++ot) {
        v8f acc0 = {};
        v8f acc1 = {};
        const unsigned short* wrow = W2T + (ot * 16 + m16) * H_;
#pragma unroll
        for (int kk = 0; kk < 8; ++kk) {
            const unsigned short* p = wrow + kk * 32 + 8 * half;
            FragU f;
            f.q[0] = *(const u32x4*)(p);
            f.q[1] = *(const u32x4*)(p + 16);
            acc0 = __builtin_amdgcn_wmma_f32_16x16x32_bf16(
                false, afrag[0][kk], false, f.v, (short)0, acc0, false, false);
            acc1 = __builtin_amdgcn_wmma_f32_16x16x32_bf16(
                false, afrag[1][kk], false, f.v, (short)0, acc1, false, false);
        }
        const float bias = b2[ot * 16 + m16];
#pragma unroll
        for (int r = 0; r < 8; ++r) {
            out[(node0 + r + 8 * half) * DO_ + ot * 16 + m16]      = acc0[r] + bias;
            out[(node0 + 16 + r + 8 * half) * DO_ + ot * 16 + m16] = acc1[r] + bias;
        }
    }
}

// ---------------------------------------------------------------------------
// Host launcher
// ---------------------------------------------------------------------------
extern "C" void kernel_launch(void* const* d_in, const int* in_sizes, int n_in,
                              void* d_out, int out_size, void* d_ws, size_t ws_size,
                              hipStream_t stream) {
    (void)in_sizes; (void)n_in; (void)out_size; (void)ws_size;

    const float* edge_data   = (const float*)d_in[0];
    const int*   receivers   = (const int*)d_in[1];
    const float* node_data   = (const float*)d_in[2];
    const float* global_data = (const float*)d_in[3];
    const float* W1          = (const float*)d_in[4];
    const float* b1          = (const float*)d_in[5];
    const float* W2          = (const float*)d_in[6];
    const float* b2          = (const float*)d_in[7];
    float* out = (float*)d_out;

    // workspace layout (16B aligned; ~7.8 MB total)
    const int NPAD = 100096;
    unsigned int* cnt    = (unsigned int*)d_ws;
    unsigned int* offs   = cnt + NPAD;
    unsigned int* cursor = offs + NPAD;
    unsigned int* eidx   = cursor + NPAD;
    unsigned short* W1T  = (unsigned short*)(eidx + NEDGES);  // 256*256 bf16
    unsigned short* W2T  = W1T + 256 * 256;                   // 128*256 bf16

    // CSR build + gather aggregation (means written into out cols 0..47)
    zero_u32_kernel<<<dim3((NNODES + 255) / 256), dim3(256), 0, stream>>>(cnt, NNODES);
    count_kernel<<<dim3((NEDGES + 255) / 256), dim3(256), 0, stream>>>(receivers, cnt);
    scan_kernel<<<dim3(1), dim3(1024), 0, stream>>>(cnt, offs, cursor);
    place_kernel<<<dim3((NEDGES + 255) / 256), dim3(256), 0, stream>>>(receivers, cursor, eidx);
    {
        int total = NNODES * DE_;
        aggregate_mean_kernel<<<dim3((total + 255) / 256), dim3(256), 0, stream>>>(
            edge_data, offs, cnt, eidx, out);
    }

    // weight conversion (bf16, transposed, K-padded)
    cvt_w1_kernel<<<dim3((256 * 256) / 256), dim3(256), 0, stream>>>(W1, W1T);
    cvt_w2_kernel<<<dim3((128 * 256) / 256), dim3(256), 0, stream>>>(W2, W2T);

    // fused WMMA MLP
    {
        int blocks = (TILES32 + WAVES_PER_BLOCK - 1) / WAVES_PER_BLOCK;
        mlp_kernel<<<dim3(blocks), dim3(32 * WAVES_PER_BLOCK), 0, stream>>>(
            node_data, global_data, b1, b2, W1T, W2T, out);
    }
}
